// EditDistNeuralModelConcurrent_4930622455856
// MI455X (gfx1250) — matrix operand order repaired and checked by reading, not compiled, not run on hardware
//
#include <hip/hip_runtime.h>
#include <hip/hip_bf16.h>
#include <math.h>
#include <cstdint>

// ---------------------------------------------------------------------------
// EditDistNeuralModel for MI455X (gfx1250, wave32, WMMA + TDM).
//  - Dense GEMMs: v_wmma_f32_16x16x32_f16, 128x64 block tile, 8 waves,
//    one A fragment reused across 4 WMMA issues per wave per k-step.
//  - A/B tiles staged to LDS by the Tensor Data Mover (tensor_load_to_lds,
//    TENSORcnt-tracked, double buffered) -- the CDNA5 async path.
//  - feat@proj_w factored into ar_vec@W_top + en_vec@W_bot (34 GFLOP -> 268 MFLOP).
//  - alpha/beta DP: anti-diagonal wavefront, 128x128 table in LDS (64 KB).
// ---------------------------------------------------------------------------

typedef __attribute__((ext_vector_type(16))) _Float16 v16h;
typedef __attribute__((ext_vector_type(8)))  float    v8f;
typedef __attribute__((ext_vector_type(4)))  unsigned v4u;
typedef __attribute__((ext_vector_type(8)))  int      v8i;
typedef __attribute__((ext_vector_type(4)))  int      v4i;

constexpr int BB = 8, TT = 128, VV = 128, HH = 256, FFI = 512;
constexpr int NLAYER = 2, NHEAD = 4, DHEAD = 64;
constexpr int TOK = BB * TT;              // 1024 rows per encoder
#define NEGV (-1e30f)
#define ACT_NONE 0
#define ACT_GELU 1

// ---------------------------------------------------------------------------
// TDM 2-D tile load: builds D# group0/group1 per CDNA5 ISA 8.3/8.4 and issues
// tensor_load_to_lds. Dims/strides in elements (data_size = 4 bytes).
// ---------------------------------------------------------------------------
__device__ __forceinline__ void tdm_load_2d(unsigned lds_off, const void* gptr,
                                            unsigned tile_x, unsigned tile_y,
                                            unsigned tensor_x, unsigned tensor_y,
                                            unsigned stride_x) {
  unsigned long long ga = (unsigned long long)(uintptr_t)gptr;
  v4u g0;
  g0[0] = 1u;                                          // count=1 (user mode)
  g0[1] = lds_off;                                     // lds_addr (bytes)
  g0[2] = (unsigned)(ga & 0xFFFFFFFFu);                // global_addr[31:0]
  g0[3] = (unsigned)((ga >> 32) & 0x01FFFFFFu)         // global_addr[56:32]
          | (2u << 30);                                // type = 2 ("image")
  unsigned long long s0 = stride_x;                    // tensor_dim0_stride
  v8i g1;
  g1[0] = (int)(2u << 16);                             // data_size = 4B, mask=0
  g1[1] = (int)((tensor_x & 0xFFFFu) << 16);           // tensor_dim0[15:0]
  g1[2] = (int)((tensor_x >> 16) | ((tensor_y & 0xFFFFu) << 16));
  g1[3] = (int)((tensor_y >> 16) | (tile_x << 16));    // tile_dim0
  g1[4] = (int)tile_y;                                 // tile_dim1 (tile_dim2=0)
  g1[5] = (int)(s0 & 0xFFFFFFFFu);                     // stride0[31:0]
  g1[6] = (int)((s0 >> 32) & 0xFFFFu);                 // stride0[47:32], stride1=0
  g1[7] = 0;
  v4i z4 = {0, 0, 0, 0};
#if defined(__clang_major__) && (__clang_major__ >= 23)
  v8i z8 = {0, 0, 0, 0, 0, 0, 0, 0};
  __builtin_amdgcn_tensor_load_to_lds(g0, g1, z4, z4, z8, 0);
#else
  __builtin_amdgcn_tensor_load_to_lds(g0, g1, z4, z4, 0);
#endif
}

// The TDM writes LDS through a descriptor (integer offset), which LLVM's alias
// analysis cannot see: without this, loads from the staging buffer fold to
// undef and the fragment code is deleted (observed in round 2 disasm).
// Escaping the LDS pointer into an empty asm with a memory clobber forces the
// compiler to treat the buffer as written. Zero runtime cost.
__device__ __forceinline__ void tdm_publish(float* lds_base) {
  asm volatile("" : "+v"(lds_base) : : "memory");
}

// ---------------------------------------------------------------------------
// Embedding + LayerNorm: one block (256 threads = H) per token.
// ---------------------------------------------------------------------------
__global__ void embed_ln_kernel(const int* __restrict__ ids,
                                const float* __restrict__ wemb,
                                const float* __restrict__ pemb,
                                const float* __restrict__ temb,
                                const float* __restrict__ g,
                                const float* __restrict__ bta,
                                float* __restrict__ out) {
  int tok = blockIdx.x;
  int s   = tok % TT;
  int h   = threadIdx.x;
  int id  = ids[tok];
  float x = wemb[id * HH + h] + pemb[s * HH + h] + temb[h];

  __shared__ float red[HH];
  red[h] = x; __syncthreads();
  for (int off = HH / 2; off > 0; off >>= 1) { if (h < off) red[h] += red[h + off]; __syncthreads(); }
  float mu = red[0] / (float)HH; __syncthreads();
  float d = x - mu;
  red[h] = d * d; __syncthreads();
  for (int off = HH / 2; off > 0; off >>= 1) { if (h < off) red[h] += red[h + off]; __syncthreads(); }
  float var = red[0] / (float)HH;
  out[tok * HH + h] = d * rsqrtf(var + 1e-12f) * g[h] + bta[h];
}

// ---------------------------------------------------------------------------
// Residual + LayerNorm (in place on x): x = LN(x + dlt) * g + b
// ---------------------------------------------------------------------------
__global__ void ln_res_kernel(float* __restrict__ x,
                              const float* __restrict__ dlt,
                              const float* __restrict__ g,
                              const float* __restrict__ bta) {
  int tok = blockIdx.x;
  int h = threadIdx.x;
  float y = x[tok * HH + h] + dlt[tok * HH + h];
  __shared__ float red[HH];
  red[h] = y; __syncthreads();
  for (int off = HH / 2; off > 0; off >>= 1) { if (h < off) red[h] += red[h + off]; __syncthreads(); }
  float mu = red[0] / (float)HH; __syncthreads();
  float d = y - mu;
  red[h] = d * d; __syncthreads();
  for (int off = HH / 2; off > 0; off >>= 1) { if (h < off) red[h] += red[h + off]; __syncthreads(); }
  float var = red[0] / (float)HH;
  x[tok * HH + h] = d * rsqrtf(var + 1e-12f) * g[h] + bta[h];
}

// ---------------------------------------------------------------------------
// WMMA GEMM with TDM double-buffered LDS staging.
//   Block tile: 128 (M) x 64 (N), 256 threads = 8 waves.
//   Wave w owns rows 16w..16w+15 and all 64 columns: 4 accumulators,
//   one A fragment -> 4 WMMA issues per k-step (K stepped by 32).
//   Wave 0 drives tensor_load_to_lds; completion via s_wait_tensorcnt.
// Fragment layouts per CDNA5 ISA 7.12.2 (16-bit A 16x32, B 32x16, f32 C 16x16).
// ---------------------------------------------------------------------------
constexpr int BM = 128, BN = 64, BK = 32;
constexpr int A_TILE = BM * BK;           // floats
constexpr int B_TILE = BK * BN;

__global__ void gemm_wmma_kernel(const float* __restrict__ A,
                                 const float* __restrict__ Bm,
                                 const float* __restrict__ bias,
                                 float* __restrict__ C,
                                 int M, int N, int K, int act) {
  // [ A buf0 | A buf1 | B buf0 | B buf1 ]  (sole static LDS object -> offset 0)
  __shared__ float smem[2 * A_TILE + 2 * B_TILE];

  int bn = blockIdx.x % (N / BN);
  int bm = blockIdx.x / (N / BN);
  int rowBase = bm * BM, colBase = bn * BN;

  int w    = threadIdx.x >> 5;            // wave 0..7
  int lane = threadIdx.x & 31;
  int l = lane & 15, g = lane >> 4;

  bool issuer = (threadIdx.x < 32);       // wave 0 drives the TDM
  int steps = K / BK;

  if (issuer) {
    tdm_load_2d(0u,                         A + (size_t)rowBase * K, BK, BM, K, M, K);
    tdm_load_2d((unsigned)(2 * A_TILE * 4), Bm + colBase,            BN, BK, N, K, N);
  }

  v8f acc[4] = {};
  for (int s = 0; s < steps; ++s) {
    int cur = s & 1;
    if (issuer) {
      if (s + 1 < steps) {
        int k0 = (s + 1) * BK;
        int nxt = cur ^ 1;
        tdm_load_2d((unsigned)(nxt * A_TILE * 4),
                    A + (size_t)rowBase * K + k0, BK, BM, K, M, K);
        tdm_load_2d((unsigned)((2 * A_TILE + nxt * B_TILE) * 4),
                    Bm + (size_t)k0 * N + colBase, BN, BK, N, K, N);
        __builtin_amdgcn_s_wait_tensorcnt((short)2);   // current buffers landed
      } else {
        __builtin_amdgcn_s_wait_tensorcnt((short)0);
      }
    }
    __syncthreads();
    tdm_publish(smem);                    // make TDM's LDS writes visible to AA

    int offA = cur * A_TILE;
    int offB = 2 * A_TILE + cur * B_TILE;

    // A fragment: lane row = 16w + l; halves 0..7 -> K=8g+0..7,
    //             halves 8..15 -> K=16+8g+0..7   (ISA 16-bit A 16x32)
    v16h a;
    int abase = offA + (w * 16 + l) * BK + g * 8;
#pragma unroll
    for (int i = 0; i < 8; ++i) a[i] = (_Float16)smem[abase + i];
#pragma unroll
    for (int i = 0; i < 8; ++i) a[8 + i] = (_Float16)smem[abase + 16 + i];

#pragma unroll
    for (int c = 0; c < 4; ++c) {
      // B fragment: lane column = 16c + l; halves 0..7 -> K=8g+0..7, 8..15 -> K=16+8g+0..7
      v16h b;
      int bbase = offB + (g * 8) * BN + c * 16 + l;
#pragma unroll
      for (int i = 0; i < 16; ++i) b[i] = (_Float16)smem[bbase + (i + (i >= 8 ? 8 : 0)) * BN];
      acc[c] = __builtin_amdgcn_wmma_f32_16x16x32_f16(
          false, a, false, b, (short)0, acc[c], false, false);
    }
    __syncthreads();
  }

#pragma unroll
  for (int c = 0; c < 4; ++c) {
    int col = colBase + c * 16 + l;
    float bv = bias ? bias[col] : 0.0f;
#pragma unroll
    for (int r = 0; r < 8; ++r) {
      float vv = acc[c][r] + bv;                 // D: VGPR r -> row 16w + r + 8g
      if (act == ACT_GELU) vv = 0.5f * vv * (1.0f + erff(vv * 0.70710678f));
      C[(size_t)(rowBase + w * 16 + r + g * 8) * N + col] = vv;
    }
  }
}

// ---------------------------------------------------------------------------
// Attention: one block per (b, head, query t). 128 threads (one per key j).
// ---------------------------------------------------------------------------
__global__ void attn_kernel(const float* __restrict__ Q,
                            const float* __restrict__ Kb,
                            const float* __restrict__ Vb,
                            float* __restrict__ ctx) {
  int t  = blockIdx.x % TT;
  int hd = (blockIdx.x / TT) % NHEAD;
  int b  = blockIdx.x / (TT * NHEAD);
  int j  = threadIdx.x;

  __shared__ float qs[DHEAD];
  __shared__ float att[TT];
  __shared__ float red[TT];

  if (j < DHEAD) qs[j] = Q[(b * TT + t) * HH + hd * DHEAD + j];
  __syncthreads();

  const float* kr = Kb + (b * TT + j) * HH + hd * DHEAD;
  float s = 0.0f;
#pragma unroll 8
  for (int d = 0; d < DHEAD; ++d) s += qs[d] * kr[d];
  s *= 0.125f;                          // 1/sqrt(64)

  red[j] = s; __syncthreads();
  for (int off = TT / 2; off > 0; off >>= 1) { if (j < off) red[j] = fmaxf(red[j], red[j + off]); __syncthreads(); }
  float m = red[0]; __syncthreads();
  float e = expf(s - m);
  att[j] = e; red[j] = e; __syncthreads();
  for (int off = TT / 2; off > 0; off >>= 1) { if (j < off) red[j] += red[j + off]; __syncthreads(); }
  float inv = 1.0f / red[0];
  __syncthreads();

  if (j < DHEAD) {
    float acc = 0.0f;
    for (int kk = 0; kk < TT; ++kk)
      acc += att[kk] * Vb[(b * TT + kk) * HH + hd * DHEAD + j];
    ctx[(b * TT + t) * HH + hd * DHEAD + j] = acc * inv;
  }
}

// ---------------------------------------------------------------------------
// Fused scores: pre = Af[b,t] + Ef[b,v]; relu; @act_w + act_b; log_softmax(4).
// ---------------------------------------------------------------------------
__global__ void scores_kernel(const float* __restrict__ Af,
                              const float* __restrict__ Ef,
                              const float* __restrict__ actw,
                              const float* __restrict__ actb,
                              float* __restrict__ scores_out) {
  int bt = blockIdx.x;
  int b  = bt / TT;
  int v  = threadIdx.x;

  __shared__ float sA[HH];
  __shared__ float sW[HH * 4];
  sA[v] = Af[bt * HH + v];
  sA[v + VV] = Af[bt * HH + v + VV];
  for (int i = v; i < HH * 4; i += VV) sW[i] = actw[i];
  __syncthreads();

  const float* er = Ef + (b * VV + v) * HH;
  float l0 = actb[0], l1 = actb[1], l2 = actb[2], l3 = actb[3];
  for (int h = 0; h < HH; ++h) {
    float r = fmaxf(sA[h] + er[h], 0.0f);
    l0 += r * sW[h * 4 + 0];
    l1 += r * sW[h * 4 + 1];
    l2 += r * sW[h * 4 + 2];
    l3 += r * sW[h * 4 + 3];
  }
  float m = fmaxf(fmaxf(l0, l1), fmaxf(l2, l3));
  float lse = m + logf(expf(l0 - m) + expf(l1 - m) + expf(l2 - m) + expf(l3 - m));
  float* o = scores_out + (bt * VV + v) * 4;
  o[0] = l0 - lse; o[1] = l1 - lse; o[2] = l2 - lse; o[3] = l3 - lse;
}

__device__ __forceinline__ float lse3(float a, float b, float c) {
  float m = fmaxf(fmaxf(a, b), c);
  return m + logf(expf(a - m) + expf(b - m) + expf(c - m));
}

// ---------------------------------------------------------------------------
// alpha DP: anti-diagonal wavefront, whole 128x128 table in LDS (64 KB).
// ---------------------------------------------------------------------------
__global__ void alpha_kernel(const float* __restrict__ sc, float* __restrict__ alphag) {
  int b = blockIdx.x;
  int v = threadIdx.x;
  __shared__ float al[TT * VV];
  const float* S = sc + b * TT * VV * 4;

  for (int d = 0; d < TT + VV - 1; ++d) {
    int t = d - v;
    if (t >= 0 && t < TT) {
      float val;
      if (t == 0 && v == 0) {
        val = 0.0f;
      } else {
        const float* c = S + (t * VV + v) * 4;          // [del, ins, sub, -]
        float left = (v > 0)          ? c[1] + al[t * VV + v - 1]       : NEGV;
        float up   = (t > 0)          ? c[0] + al[(t - 1) * VV + v]     : NEGV;
        float diag = (t > 0 && v > 0) ? c[2] + al[(t - 1) * VV + v - 1] : NEGV;
        val = lse3(left, up, diag);
      }
      al[t * VV + v] = val;
    }
    __syncthreads();
  }
  for (int t = 0; t < TT; ++t) alphag[(b * TT + t) * VV + v] = al[t * VV + v];
}

// ---------------------------------------------------------------------------
// beta DP: reversed wavefront.
// ---------------------------------------------------------------------------
__global__ void beta_kernel(const float* __restrict__ sc, float* __restrict__ betag) {
  int b = blockIdx.x;
  int v = threadIdx.x;
  __shared__ float be[TT * VV];
  const float* S = sc + b * TT * VV * 4;

  for (int d = TT + VV - 2; d >= 0; --d) {
    int t = d - v;
    if (t >= 0 && t < TT) {
      float val;
      if (t == TT - 1 && v == VV - 1) {
        val = 0.0f;
      } else {
        float right = (v + 1 < VV) ? S[(t * VV + v + 1) * 4 + 1] + be[t * VV + v + 1] : NEGV;
        float down  = (t + 1 < TT) ? S[((t + 1) * VV + v) * 4 + 0] + be[(t + 1) * VV + v] : NEGV;
        float dd    = (t + 1 < TT && v + 1 < VV)
                        ? S[((t + 1) * VV + v + 1) * 4 + 2] + be[(t + 1) * VV + v + 1] : NEGV;
        val = lse3(right, down, dd);
      }
      be[t * VV + v] = val;
    }
    __syncthreads();
  }
  for (int t = 0; t < TT; ++t) betag[(b * TT + t) * VV + v] = be[t * VV + v];
}

// ---------------------------------------------------------------------------
// Expected counts = exp(stacked - logsumexp(stacked)), plus alpha[0,-1,-1].
// ---------------------------------------------------------------------------
__global__ void expected_kernel(const float* __restrict__ sc,
                                const float* __restrict__ alphag,
                                const float* __restrict__ betag,
                                float* __restrict__ counts,
                                float* __restrict__ scalar_out) {
  int bt = blockIdx.x;
  int b = bt / TT, t = bt % TT;
  int v = threadIdx.x;
  const float* S  = sc + (bt * VV + v) * 4;
  const float* al = alphag + b * TT * VV;
  const float* be = betag + b * TT * VV;
  const float NI = -INFINITY;

  float bv = be[t * VV + v];
  float ed = (t >= 1 && t < TT - 1) ? al[(t - 1) * VV + v] + S[0] + bv : NI;
  float ei = (v >= 1 && v < VV - 1) ? al[t * VV + v - 1] + S[1] + bv : NI;
  float es = (t >= 1 && t < TT - 1 && v >= 1 && v < VV - 1)
               ? al[(t - 1) * VV + v - 1] + S[2] + bv : NI;

  float m = fmaxf(fmaxf(ed, ei), es);
  float* o = counts + (bt * VV + v) * 4;
  if (m == NI) {
    o[0] = o[1] = o[2] = o[3] = 0.0f;   // no plausible action -> denom guard
  } else {
    float e0 = expf(ed - m), e1 = expf(ei - m), e2 = expf(es - m);
    float inv = 1.0f / (e0 + e1 + e2);  // exp(x - denom) == e_x / sum
    o[0] = e0 * inv; o[1] = e1 * inv; o[2] = e2 * inv; o[3] = 0.0f;
  }
  if (bt == 0 && v == 0)
    scalar_out[0] = alphag[(TT - 1) * VV + (VV - 1)];   // alpha[0,-1,-1]
}

// ---------------------------------------------------------------------------
// Host-side orchestration
// ---------------------------------------------------------------------------
struct BertP {
  const float *wemb, *pemb, *temb, *embg, *embb;
  const float *qw, *qb, *kw, *kb, *vw, *vb, *aow, *aob, *ag, *ab;
  const float *f1w, *f1b, *f2w, *f2b, *og, *ob;
};

static inline BertP load_params(void* const* din, int base) {
  BertP p;
  p.wemb = (const float*)din[base + 0];  p.pemb = (const float*)din[base + 1];
  p.temb = (const float*)din[base + 2];  p.embg = (const float*)din[base + 3];
  p.embb = (const float*)din[base + 4];
  p.qw = (const float*)din[base + 5];    p.qb = (const float*)din[base + 6];
  p.kw = (const float*)din[base + 7];    p.kb = (const float*)din[base + 8];
  p.vw = (const float*)din[base + 9];    p.vb = (const float*)din[base + 10];
  p.aow = (const float*)din[base + 11];  p.aob = (const float*)din[base + 12];
  p.ag = (const float*)din[base + 13];   p.ab = (const float*)din[base + 14];
  p.f1w = (const float*)din[base + 15];  p.f1b = (const float*)din[base + 16];
  p.f2w = (const float*)din[base + 17];  p.f2b = (const float*)din[base + 18];
  p.og = (const float*)din[base + 19];   p.ob = (const float*)din[base + 20];
  return p;
}

static inline void gemm(const float* A, const float* Bm, const float* bias, float* C,
                        int M, int N, int K, int act, hipStream_t st) {
  int blocks = (M / BM) * (N / BN);
  gemm_wmma_kernel<<<blocks, 256, 0, st>>>(A, Bm, bias, C, M, N, K, act);
}

static void run_encoder(const int* ids, const BertP& p, float* x,
                        float* q, float* k, float* v, float* ctx,
                        float* t1, float* h1, hipStream_t st) {
  embed_ln_kernel<<<TOK, HH, 0, st>>>(ids, p.wemb, p.pemb, p.temb, p.embg, p.embb, x);
  for (int i = 0; i < NLAYER; ++i) {
    const float *qw = p.qw + i * HH * HH, *qb = p.qb + i * HH;
    const float *kw = p.kw + i * HH * HH, *kb = p.kb + i * HH;
    const float *vw = p.vw + i * HH * HH, *vb = p.vb + i * HH;
    const float *aow = p.aow + i * HH * HH, *aob = p.aob + i * HH;
    const float *f1w = p.f1w + i * HH * FFI, *f1b = p.f1b + i * FFI;
    const float *f2w = p.f2w + i * FFI * HH, *f2b = p.f2b + i * HH;

    gemm(x, qw, qb, q, TOK, HH, HH, ACT_NONE, st);
    gemm(x, kw, kb, k, TOK, HH, HH, ACT_NONE, st);
    gemm(x, vw, vb, v, TOK, HH, HH, ACT_NONE, st);
    attn_kernel<<<BB * NHEAD * TT, TT, 0, st>>>(q, k, v, ctx);
    gemm(ctx, aow, aob, t1, TOK, HH, HH, ACT_NONE, st);
    ln_res_kernel<<<TOK, HH, 0, st>>>(x, t1, p.ag + i * HH, p.ab + i * HH);
    gemm(x, f1w, f1b, h1, TOK, FFI, HH, ACT_GELU, st);
    gemm(h1, f2w, f2b, t1, TOK, HH, FFI, ACT_NONE, st);
    ln_res_kernel<<<TOK, HH, 0, st>>>(x, t1, p.og + i * HH, p.ob + i * HH);
  }
}

extern "C" void kernel_launch(void* const* d_in, const int* in_sizes, int n_in,
                              void* d_out, int out_size, void* d_ws, size_t ws_size,
                              hipStream_t stream) {
  (void)in_sizes; (void)n_in; (void)out_size; (void)ws_size;
  // Input layout (nested dicts flattened in insertion order):
  // 0 ar_sent, 1 en_sent, 2..22 ar_params (21), 23..43 en_params (21),
  // 44 proj_w [512,256], 45 proj_b [256], 46 act_w [256,4], 47 act_b [4]
  const int* ar_ids = (const int*)d_in[0];
  const int* en_ids = (const int*)d_in[1];
  BertP arP = load_params(d_in, 2);
  BertP enP = load_params(d_in, 23);
  const float* proj_w = (const float*)d_in[44];
  const float* proj_b = (const float*)d_in[45];
  const float* act_w  = (const float*)d_in[46];
  const float* act_b  = (const float*)d_in[47];
  float* out = (float*)d_out;

  // Workspace carve-up (floats)
  float* ws = (float*)d_ws;
  float* ar_x = ws;                         // 1024*256
  float* en_x = ar_x + TOK * HH;
  float* qb   = en_x + TOK * HH;
  float* kb   = qb + TOK * HH;
  float* vb   = kb + TOK * HH;
  float* ctx  = vb + TOK * HH;
  float* t1   = ctx + TOK * HH;
  float* h1   = t1 + TOK * HH;              // 1024*512
  float* Af   = h1 + TOK * FFI;
  float* Ef   = Af + TOK * HH;
  float* alphag = Ef + TOK * HH;            // 8*128*128
  float* betag  = alphag + BB * TT * VV;

  run_encoder(ar_ids, arP, ar_x, qb, kb, vb, ctx, t1, h1, stream);
  run_encoder(en_ids, enP, en_x, qb, kb, vb, ctx, t1, h1, stream);

  // Factored projection: feat@proj_w == ar_x@W_top (+b) broadcast + en_x@W_bot broadcast
  gemm(ar_x, proj_w,           proj_b,  Af, TOK, HH, HH, ACT_NONE, stream);
  gemm(en_x, proj_w + HH * HH, nullptr, Ef, TOK, HH, HH, ACT_NONE, stream);

  scores_kernel<<<BB * TT, VV, 0, stream>>>(Af, Ef, act_w, act_b, out);
  alpha_kernel<<<BB, VV, 0, stream>>>(out, alphag);
  beta_kernel<<<BB, VV, 0, stream>>>(out, betag);

  const int BTV4 = BB * TT * VV * 4;
  expected_kernel<<<BB * TT, VV, 0, stream>>>(out, alphag, betag,
                                              out + BTV4, out + 2 * BTV4);
}
